// VQVAEModel_45140106281595
// MI455X (gfx1250) — matrix operand order repaired and checked by reading, not compiled, not run on hardware
//
// VQ-VAE forward for MI455X (gfx1250): bf16 WMMA GEMMs for all dense/conv-as-GEMM
// work, async-to-LDS double-buffered staging, VALU kernels for attention / VQ /
// groupnorm / deconvs.  Workspace: ~224 MB (bump-allocated, big region reused).
#include <hip/hip_runtime.h>

typedef unsigned short u16;
typedef __attribute__((ext_vector_type(16))) __bf16 v16bf;
typedef __attribute__((ext_vector_type(8)))  float  v8f;

#ifndef __has_builtin
#define __has_builtin(x) 0
#endif
#if __has_builtin(__builtin_amdgcn_global_load_async_to_lds_b128) && \
    __has_builtin(__builtin_amdgcn_s_wait_asynccnt)
#define USE_ASYNC 1
#else
#define USE_ASYNC 0
#endif

__device__ __forceinline__ u16 f2bf(float f) {
  unsigned int u = __float_as_uint(f);
  u += 0x7FFFu + ((u >> 16) & 1u);          // round-to-nearest-even
  return (u16)(u >> 16);
}

union FragBF { uint4 q[2]; v16bf v; };

#if USE_ASYNC
// builtin signature (from clang diagnostic): (v4i AS1*, v4i AS3*, imm int, imm int)
typedef int v4i_t __attribute__((vector_size(16)));
typedef __attribute__((address_space(1))) v4i_t GV4;
typedef __attribute__((address_space(3))) v4i_t LV4;
// 16B global -> LDS direct copy, tracked by ASYNCcnt (no VGPR staging).
__device__ __forceinline__ void async_cp16(const u16* g, const u16* l) {
  __builtin_amdgcn_global_load_async_to_lds_b128(
      (GV4*)(unsigned long long)g,
      (LV4*)(unsigned)(unsigned long long)l, 0, 0);
}
#endif

__device__ __forceinline__ void wait_async0() {
#if USE_ASYNC
  __builtin_amdgcn_s_wait_asynccnt(0);
#endif
}

// ---------------------------------------------------------------------------
// Tiled WMMA GEMM:  C[M,N] = act( A[M,K](bf16) * B[N,K]^T(bf16) + bias )
// 256 threads = 8 waves (2x4), block tile 128x128, K-step 32, double-buffered
// async global->LDS staging.  M,K multiples of 128/32 guaranteed by caller;
// N masked when < 128 tile (masked tiles use the zero-filling sync copy).
// act: 0=none, 1=leaky-relu(0.2), 2=relu.  C (f32) and Cb (bf16) nullable.
// ---------------------------------------------------------------------------
__global__ __launch_bounds__(256) void k_gemm(
    const u16* __restrict__ A, const u16* __restrict__ Bw,
    const float* __restrict__ bias,
    float* __restrict__ C, u16* __restrict__ Cb,
    int M, int N, int K, int act)
{
  // 32 bf16 + pad to 80B rows (16B aligned, conflict-free b128), x2 buffers
  __shared__ u16 sA[2 * 128 * 40];
  __shared__ u16 sB[2 * 128 * 40];

  const int tid  = threadIdx.x;
  const int lane = tid & 31;
  const int wave = tid >> 5;
  const int wm   = wave >> 2;          // 0..1  (64 rows per wave)
  const int wn   = wave & 3;           // 0..3  (32 cols per wave)
  const int bm   = blockIdx.y * 128;
  const int bn   = blockIdx.x * 128;
  const int nk   = K >> 5;
  const bool fullB = (bn + 128 <= N);

  v8f acc[4][2];
#pragma unroll
  for (int i = 0; i < 4; i++)
#pragma unroll
    for (int j = 0; j < 2; j++)
#pragma unroll
      for (int e = 0; e < 8; e++) acc[i][j][e] = 0.f;

  // stage K-tile t into LDS buffer buf (each thread: 2 x 16B for A, 2 for B)
  auto stage = [&](int t, int buf) {
    const int k0 = t << 5;
    u16* dA = sA + buf * (128 * 40);
    u16* dB = sB + buf * (128 * 40);
#pragma unroll
    for (int s = 0; s < 2; s++) {
      int seg = tid + (s << 8);
      int r   = seg >> 2;
      int cs  = (seg & 3) << 3;
      const u16* ga = A + (size_t)(bm + r) * K + k0 + cs;
#if USE_ASYNC
      async_cp16(ga, &dA[r * 40 + cs]);
#else
      *(uint4*)&dA[r * 40 + cs] = *(const uint4*)ga;
#endif
      int nb = bn + r;
      const u16* gb = Bw + (size_t)nb * K + k0 + cs;
      if (fullB) {
#if USE_ASYNC
        async_cp16(gb, &dB[r * 40 + cs]);
#else
        *(uint4*)&dB[r * 40 + cs] = *(const uint4*)gb;
#endif
      } else {
        uint4 bv = make_uint4(0u, 0u, 0u, 0u);
        if (nb < N) bv = *(const uint4*)gb;
        *(uint4*)&dB[r * 40 + cs] = bv;
      }
      // speculative prefetch two K-tiles ahead (global_prefetch_b8)
      if (t + 2 < nk) {
        __builtin_prefetch(ga + 64, 0, 1);
        if (fullB) __builtin_prefetch(gb + 64, 0, 1);
      }
    }
  };

  stage(0, 0);
  wait_async0();
  __syncthreads();

  for (int t = 0; t < nk; t++) {
    const int cur = t & 1;
    if (t + 1 < nk) stage(t + 1, cur ^ 1);   // overlap copy of next tile

    const u16* pA = sA + cur * (128 * 40);
    const u16* pB = sB + cur * (128 * 40);
    FragBF a[4], b[2];
    const int ah = (lane >> 4) << 3;         // A: k-halves 0/8 (+16)
#pragma unroll
    for (int mi = 0; mi < 4; mi++) {
      int r = wm * 64 + mi * 16 + (lane & 15);
      a[mi].q[0] = *(const uint4*)&pA[r * 40 + ah];
      a[mi].q[1] = *(const uint4*)&pA[r * 40 + ah + 16];
    }
    const int bh = (lane >> 4) << 3;         // B: k-halves 0/8 (+16)
#pragma unroll
    for (int ni = 0; ni < 2; ni++) {
      int n = wn * 32 + ni * 16 + (lane & 15);
      b[ni].q[0] = *(const uint4*)&pB[n * 40 + bh];
      b[ni].q[1] = *(const uint4*)&pB[n * 40 + bh + 16];
    }
#pragma unroll
    for (int mi = 0; mi < 4; mi++)
#pragma unroll
      for (int ni = 0; ni < 2; ni++)
        acc[mi][ni] = __builtin_amdgcn_wmma_f32_16x16x32_bf16(
            false, a[mi].v, false, b[ni].v, (short)0, acc[mi][ni], false, false);

    wait_async0();        // next tile resident before anyone signals
    __syncthreads();
  }

  const int nlo = lane & 15;
  const int mhi = (lane >> 4) << 3;
#pragma unroll
  for (int mi = 0; mi < 4; mi++) {
#pragma unroll
    for (int ni = 0; ni < 2; ni++) {
      int col = bn + wn * 32 + ni * 16 + nlo;
      if (col >= N) continue;
      float bv = bias ? bias[col] : 0.f;
#pragma unroll
      for (int r = 0; r < 8; r++) {
        int row = bm + wm * 64 + mi * 16 + r + mhi;
        float v = acc[mi][ni][r] + bv;
        if (act == 1)      v = (v >= 0.f) ? v : 0.2f * v;
        else if (act == 2) v = fmaxf(v, 0.f);
        size_t o = (size_t)row * N + col;
        if (C)  C[o]  = v;
        if (Cb) Cb[o] = f2bf(v);
      }
    }
  }
}

// ---------------------------------------------------------------------------
// f32 -> bf16 convert with optional zero-pad of inner dim (K -> Kp)
// ---------------------------------------------------------------------------
__global__ void k_cvt(const float* __restrict__ src, u16* __restrict__ dst,
                      long rows, long K, long Kp) {
  long i = (long)blockIdx.x * 256 + threadIdx.x;
  if (i >= rows * Kp) return;
  long r = i / Kp, c = i - r * Kp;
  dst[i] = (c < K) ? f2bf(src[r * K + c]) : (u16)0;
}

// OIHW [Cout][Cin][3][3] f32 -> bf16 [Cout][tap*Cin+ci]  (matches im2col K order)
__global__ void k_repack3(const float* __restrict__ w, u16* __restrict__ dst,
                          int Cout, int Cin) {
  int i = blockIdx.x * 256 + threadIdx.x;
  if (i >= Cout * Cin * 9) return;
  int co  = i / (Cin * 9);
  int rem = i - co * Cin * 9;
  int tap = rem / Cin;
  int ci  = rem - tap * Cin;
  dst[i] = f2bf(w[((size_t)(co * Cin + ci)) * 9 + tap]);
}

// ---------------------------------------------------------------------------
// CBAM channel attention: per batch, per-channel max/mean -> 128->8->128 MLP
// ---------------------------------------------------------------------------
__global__ void k_chan_att(const float* __restrict__ z, const float* __restrict__ w1,
                           const float* __restrict__ w2, float* __restrict__ cw) {
  int b = blockIdx.x, c = threadIdx.x;          // 512 blocks x 128 threads
  const float* zp = z + ((size_t)b * 128 + c) * 64;
  float mx = -3.4e38f, sm = 0.f;
  for (int p = 0; p < 64; p++) { float v = zp[p]; sm += v; mx = fmaxf(mx, v); }
  __shared__ float smax[128], smean[128], hid[16];
  smax[c] = mx; smean[c] = sm * (1.f / 64.f);
  __syncthreads();
  if (c < 16) {
    int r = c & 7;
    const float* src = (c < 8) ? smax : smean;
    float s = 0.f;
    for (int j = 0; j < 128; j++) s += w1[r * 128 + j] * src[j];
    hid[c] = fmaxf(s, 0.f);
  }
  __syncthreads();
  float s = 0.f;
  for (int r = 0; r < 8; r++) s += w2[c * 8 + r] * (hid[r] + hid[8 + r]);
  cw[b * 128 + c] = 1.f / (1.f + expf(-s));
}

// sfeat[b][0]=mean_c, sfeat[b][1]=max_c of (z*cw)
__global__ void k_sfeat(const float* __restrict__ z, const float* __restrict__ cw,
                        float* __restrict__ sf) {
  int i = blockIdx.x * 256 + threadIdx.x;
  if (i >= 512 * 64) return;
  int b = i >> 6, p = i & 63;
  float sm = 0.f, mx = -3.4e38f;
  for (int c = 0; c < 128; c++) {
    float v = z[((size_t)b * 128 + c) * 64 + p] * cw[b * 128 + c];
    sm += v; mx = fmaxf(mx, v);
  }
  sf[(size_t)b * 128 + p]      = sm * (1.f / 128.f);
  sf[(size_t)b * 128 + 64 + p] = mx;
}

// 7x7 pad-3 conv on 8x8, 2ch -> 1ch, sigmoid
__global__ void k_spat(const float* __restrict__ sf, const float* __restrict__ sw,
                       float* __restrict__ sa) {
  int i = blockIdx.x * 256 + threadIdx.x;
  if (i >= 512 * 64) return;
  int b = i >> 6, p = i & 63;
  int oy = p >> 3, ox = p & 7;
  float s = 0.f;
  for (int ky = 0; ky < 7; ky++) {
    int iy = oy + ky - 3;
    if ((unsigned)iy >= 8u) continue;
    for (int kx = 0; kx < 7; kx++) {
      int ix = ox + kx - 3;
      if ((unsigned)ix >= 8u) continue;
      int q = iy * 8 + ix;
      s += sw[ky * 7 + kx]      * sf[(size_t)b * 128 + q]
         + sw[49 + ky * 7 + kx] * sf[(size_t)b * 128 + 64 + q];
    }
  }
  sa[i] = 1.f / (1.f + expf(-s));
}

// z_att (NCHW f32) -> NHWC bf16 [32768,128], both attentions applied
__global__ void k_apply(const float* __restrict__ z, const float* __restrict__ cw,
                        const float* __restrict__ sa, u16* __restrict__ dst) {
  int i = blockIdx.x * 256 + threadIdx.x;
  if (i >= 512 * 64 * 128) return;
  int c = i & 127, m = i >> 7;
  int b = m >> 6, p = m & 63;
  float v = z[((size_t)b * 128 + c) * 64 + p] * cw[b * 128 + c] * sa[m];
  dst[(size_t)m * 128 + c] = f2bf(v);
}

// ---------------------------------------------------------------------------
// VQ helpers
// ---------------------------------------------------------------------------
__global__ void k_cnorm(const float* __restrict__ cb, float* __restrict__ cn) {
  int j = blockIdx.x * 256 + threadIdx.x;
  if (j >= 512) return;
  float s = 0.f;
  for (int c = 0; c < 64; c++) { float v = cb[j * 64 + c]; s += v * v; }
  cn[j] = s;
}

__global__ void k_zero_hist(int* __restrict__ h) {
  int i = blockIdx.x * 256 + threadIdx.x;
  if (i < 512) h[i] = 0;
}

__global__ void k_argmin(const float* __restrict__ dot, const float* __restrict__ cn,
                         int* __restrict__ idx, int* __restrict__ hist) {
  int i = blockIdx.x * 256 + threadIdx.x;
  if (i >= 32768) return;
  const float* dp = dot + (size_t)i * 512;
  float best = 3.4e38f; int bi = 0;
  for (int j = 0; j < 512; j++) {
    float d = cn[j] - 2.f * dp[j];      // ||zq||^2 constant per row: drop for argmin
    if (d < best) { best = d; bi = j; }
  }
  idx[i] = bi;
  atomicAdd(&hist[bi], 1);
}

__global__ void k_gather(const int* __restrict__ idx, const float* __restrict__ cb,
                         float* __restrict__ quant) {
  int i = blockIdx.x * 256 + threadIdx.x;
  if (i >= 32768 * 64) return;
  int r = i >> 6, c = i & 63;
  quant[i] = cb[(size_t)idx[r] * 64 + c];
}

__global__ void k_loss(const float* __restrict__ quant, const float* __restrict__ zq,
                       float* __restrict__ part) {
  __shared__ float s[256];
  int t = threadIdx.x;
  int i = blockIdx.x * 256 + t;          // exactly 8192*256 = 2,097,152 elems
  float d = quant[i] - zq[i];
  s[t] = d * d;
  __syncthreads();
  for (int st = 128; st > 0; st >>= 1) {
    if (t < st) s[t] += s[t + st];
    __syncthreads();
  }
  if (t == 0) part[blockIdx.x] = s[0];
}

__global__ void k_loss_final(const float* __restrict__ part, float* __restrict__ out0) {
  __shared__ float s[256];
  int t = threadIdx.x;
  float a = 0.f;
  for (int i = t; i < 8192; i += 256) a += part[i];
  s[t] = a;
  __syncthreads();
  for (int st = 128; st > 0; st >>= 1) {
    if (t < st) s[t] += s[t + st];
    __syncthreads();
  }
  if (t == 0) out0[0] = 1.25f * s[0] / 2097152.f;   // codebook + 0.25*encoder MSE
}

__global__ void k_perp(const int* __restrict__ hist, float* __restrict__ outp) {
  __shared__ float s[512];
  int t = threadIdx.x;
  float a = (float)hist[t] / 32768.f;
  s[t] = a * logf(a + 1e-10f);
  __syncthreads();
  for (int st = 256; st > 0; st >>= 1) {
    if (t < st) s[t] += s[t + st];
    __syncthreads();
  }
  if (t == 0) outp[0] = expf(-s[0]);
}

// ---------------------------------------------------------------------------
// im2col for 3x3 pad-1 conv on 8x8 NHWC input -> bf16 [32768, Cin*9]
// ---------------------------------------------------------------------------
__global__ void k_im2col3(const float* __restrict__ src, u16* __restrict__ dst,
                          int Cin, int relu) {
  long i = (long)blockIdx.x * 256 + threadIdx.x;
  long tot = (long)32768 * Cin * 9;
  if (i >= tot) return;
  int Krow = Cin * 9;
  long m = i / Krow;
  int  k = (int)(i - m * Krow);
  int tap = k / Cin, ci = k - tap * Cin;
  int ky = tap / 3, kx = tap - ky * 3;
  int b = (int)(m >> 6), p = (int)(m & 63);
  int iy = (p >> 3) + ky - 1, ix = (p & 7) + kx - 1;
  float v = 0.f;
  if ((unsigned)iy < 8u && (unsigned)ix < 8u) {
    v = src[(((size_t)b * 64) + iy * 8 + ix) * Cin + ci];
    if (relu) v = fmaxf(v, 0.f);
  }
  dst[i] = f2bf(v);
}

// GroupNorm C=32 groups=32 (per-channel over 64 spatial) + relu, bf16 out
__global__ void k_gn1(const float* __restrict__ x, const float* __restrict__ g,
                      const float* __restrict__ bt, u16* __restrict__ out) {
  int b = blockIdx.x >> 5, ch = blockIdx.x & 31;
  int p = threadIdx.x;                    // 64
  size_t o = ((size_t)b * 64 + p) * 32 + ch;
  float v = x[o];
  __shared__ float s1[64], s2[64];
  s1[p] = v; s2[p] = v * v;
  __syncthreads();
  for (int st = 32; st > 0; st >>= 1) {
    if (p < st) { s1[p] += s1[p + st]; s2[p] += s2[p + st]; }
    __syncthreads();
  }
  float mean = s1[0] * (1.f / 64.f);
  float var  = s2[0] * (1.f / 64.f) - mean * mean;
  float xn = (v - mean) * rsqrtf(var + 1e-5f);
  out[o] = f2bf(fmaxf(xn * g[ch] + bt[ch], 0.f));
}

// GroupNorm C=128 groups=32 (4ch x 64sp per group), add into residual stream
__global__ void k_gn2add(const float* __restrict__ x, const float* __restrict__ g,
                         const float* __restrict__ bt, float* __restrict__ h) {
  int b = blockIdx.x >> 5, grp = blockIdx.x & 31;
  int t = threadIdx.x;                    // 256
  int ch = grp * 4 + (t >> 6);
  int p  = t & 63;
  size_t o = ((size_t)b * 64 + p) * 128 + ch;
  float v = x[o];
  __shared__ float s1[256], s2[256];
  s1[t] = v; s2[t] = v * v;
  __syncthreads();
  for (int st = 128; st > 0; st >>= 1) {
    if (t < st) { s1[t] += s1[t + st]; s2[t] += s2[t + st]; }
    __syncthreads();
  }
  float mean = s1[0] * (1.f / 256.f);
  float var  = s2[0] * (1.f / 256.f) - mean * mean;
  float xn = (v - mean) * rsqrtf(var + 1e-5f);
  h[o] += xn * g[ch] + bt[ch];
}

// ---------------------------------------------------------------------------
// ConvTranspose2d k=4 s=2 p=1:  out[oy,ox,co] = sum w_t[ci,co,ky,kx]*in[iy,ix,ci]
// with oy = 2*iy - 1 + ky.  deconv1: 8x8x128 -> 16x16x64, relu in+out.
// ---------------------------------------------------------------------------
__global__ void k_deconv1(const float* __restrict__ h, const float* __restrict__ w,
                          const float* __restrict__ bias, float* __restrict__ out) {
  int i = blockIdx.x * 256 + threadIdx.x;
  if (i >= 512 * 16 * 16 * 64) return;
  int co = i & 63;
  int p  = i >> 6;
  int ox = p & 15; p >>= 4;
  int oy = p & 15; int b = p >> 4;
  float s = bias[co];
  for (int ky = 0; ky < 4; ky++) {
    int t = oy + 1 - ky;
    if (t < 0 || (t & 1)) continue;
    int iy = t >> 1; if (iy >= 8) continue;
    for (int kx = 0; kx < 4; kx++) {
      int u = ox + 1 - kx;
      if (u < 0 || (u & 1)) continue;
      int ix = u >> 1; if (ix >= 8) continue;
      const float* ip = h + ((size_t)(b * 64) + iy * 8 + ix) * 128;
      int tap = ky * 4 + kx;
      float acc = 0.f;
      for (int ci = 0; ci < 128; ci++)
        acc += fmaxf(ip[ci], 0.f) * w[(size_t)ci * 1024 + co * 16 + tap];
      s += acc;
    }
  }
  out[i] = fmaxf(s, 0.f);
}

// deconv2: 16x16x64 -> 32x32x3, NCHW output written straight into d_out+1
__global__ void k_deconv2(const float* __restrict__ h, const float* __restrict__ w,
                          const float* __restrict__ bias, float* __restrict__ out) {
  int i = blockIdx.x * 256 + threadIdx.x;
  if (i >= 512 * 3 * 32 * 32) return;
  int ox = i & 31; int p = i >> 5;
  int oy = p & 31; p >>= 5;
  int co = p % 3;  int b = p / 3;
  float s = bias[co];
  for (int ky = 0; ky < 4; ky++) {
    int t = oy + 1 - ky;
    if (t < 0 || (t & 1)) continue;
    int iy = t >> 1; if (iy >= 16) continue;
    for (int kx = 0; kx < 4; kx++) {
      int u = ox + 1 - kx;
      if (u < 0 || (u & 1)) continue;
      int ix = u >> 1; if (ix >= 16) continue;
      const float* ip = h + ((size_t)(b * 256) + iy * 16 + ix) * 64;
      int tap = ky * 4 + kx;
      float acc = 0.f;
      for (int ci = 0; ci < 64; ci++)
        acc += ip[ci] * w[(size_t)ci * 48 + co * 16 + tap];
      s += acc;
    }
  }
  out[i] = s;
}

// ---------------------------------------------------------------------------
extern "C" void kernel_launch(void* const* d_in, const int* in_sizes, int n_in,
                              void* d_out, int out_size, void* d_ws, size_t ws_size,
                              hipStream_t stream) {
  (void)in_sizes; (void)n_in; (void)ws_size;
  const float* x           = (const float*)d_in[0];
  const float* enc_w1      = (const float*)d_in[1];
  const float* enc_b1      = (const float*)d_in[2];
  const float* enc_w2      = (const float*)d_in[3];
  const float* enc_b2      = (const float*)d_in[4];
  const float* enc_w3      = (const float*)d_in[5];
  const float* enc_b3      = (const float*)d_in[6];
  const float* enc_w4      = (const float*)d_in[7];
  const float* enc_b4      = (const float*)d_in[8];
  const float* ca_w1       = (const float*)d_in[9];
  const float* ca_w2       = (const float*)d_in[10];
  const float* sa_w        = (const float*)d_in[11];
  const float* pre_w       = (const float*)d_in[12];
  const float* pre_b       = (const float*)d_in[13];
  const float* codebook    = (const float*)d_in[14];
  const float* dec_conv1_w = (const float*)d_in[15];
  const float* dec_conv1_b = (const float*)d_in[16];
  const float* deconv1_w   = (const float*)d_in[17];
  const float* deconv1_b   = (const float*)d_in[18];
  const float* deconv2_w   = (const float*)d_in[19];
  const float* deconv2_b   = (const float*)d_in[20];
  const float* resp[12];
  for (int j = 0; j < 12; j++) resp[j] = (const float*)d_in[21 + j];

  float* out = (float*)d_out;

  // ---- bump allocator over workspace --------------------------------------
  char* ws = (char*)d_ws;
  size_t off = 0;
  auto alloc = [&](size_t bytes) -> void* {
    off = (off + 255) & ~(size_t)255;
    void* p = ws + off;
    off += bytes;
    return p;
  };
  // BIG is time-shared: bf16 weight staging (<=74MB) -> VQ dot f32 (67MB) -> im2col (<=75.5MB)
  char*  big   = (char*) alloc((size_t)76 * 1024 * 1024);
  u16*   xb    = (u16*)  alloc((size_t)512 * 9024 * 2);
  u16*   h1b   = (u16*)  alloc((size_t)512 * 4096 * 2);
  u16*   h2b   = (u16*)  alloc((size_t)512 * 2048 * 2);
  u16*   h3b   = (u16*)  alloc((size_t)512 * 4096 * 2);
  float* z     = (float*)alloc((size_t)512 * 8192 * 4);
  float* cw    = (float*)alloc((size_t)512 * 128 * 4);
  float* sf    = (float*)alloc((size_t)512 * 128 * 4);
  float* sa    = (float*)alloc((size_t)512 * 64 * 4);
  u16*   znb   = (u16*)  alloc((size_t)32768 * 128 * 2);
  u16*   prewb = (u16*)  alloc((size_t)64 * 128 * 2);
  float* zq    = (float*)alloc((size_t)32768 * 64 * 4);
  u16*   zqb   = (u16*)  alloc((size_t)32768 * 64 * 2);
  u16*   cbb   = (u16*)  alloc((size_t)512 * 64 * 2);
  float* cn    = (float*)alloc((size_t)512 * 4);
  int*   vidx  = (int*)  alloc((size_t)32768 * 4);
  int*   hist  = (int*)  alloc((size_t)512 * 4);
  float* part  = (float*)alloc((size_t)8192 * 4);
  float* quant = (float*)alloc((size_t)32768 * 64 * 4);
  float* hdec  = (float*)alloc((size_t)32768 * 128 * 4);
  float* rbuf  = (float*)alloc((size_t)32768 * 32 * 4);
  u16*   rbufb = (u16*)  alloc((size_t)32768 * 32 * 2);
  float* r2    = (float*)alloc((size_t)32768 * 128 * 4);
  u16*   dcwb  = (u16*)  alloc((size_t)128 * 576 * 2);
  u16*   rw1b  = (u16*)  alloc((size_t)32 * 1152 * 2);
  u16*   rw2b  = (u16*)  alloc((size_t)128 * 32 * 2);
  float* dd1   = (float*)alloc((size_t)512 * 16 * 16 * 64 * 4);

  auto cvt = [&](const float* s, u16* d, long rows, long K, long Kp) {
    long tot = rows * Kp;
    k_cvt<<<(unsigned)((tot + 255) / 256), 256, 0, stream>>>(s, d, rows, K, Kp);
  };
  auto gemm = [&](const u16* A, const u16* Bw, const float* bias,
                  float* C, u16* Cb, int M, int N, int K, int act) {
    dim3 g((N + 127) / 128, M / 128);
    k_gemm<<<g, 256, 0, stream>>>(A, Bw, bias, C, Cb, M, N, K, act);
  };

  // ---- encoder MLP (bf16 WMMA, K of layer 1 zero-padded 9000->9024) -------
  cvt(x,      xb,         512,  9000, 9024);
  cvt(enc_w1, (u16*)big, 4096,  9000, 9024);
  gemm(xb,  (u16*)big, enc_b1, nullptr, h1b, 512, 4096, 9024, 1);
  cvt(enc_w2, (u16*)big, 2048,  4096, 4096);
  gemm(h1b, (u16*)big, enc_b2, nullptr, h2b, 512, 2048, 4096, 1);
  cvt(enc_w3, (u16*)big, 4096,  2048, 2048);
  gemm(h2b, (u16*)big, enc_b3, nullptr, h3b, 512, 4096, 2048, 1);
  cvt(enc_w4, (u16*)big, 8192,  4096, 4096);
  gemm(h3b, (u16*)big, enc_b4, z, nullptr, 512, 8192, 4096, 0);

  // ---- CBAM attention -----------------------------------------------------
  k_chan_att<<<512, 128, 0, stream>>>(z, ca_w1, ca_w2, cw);
  k_sfeat<<<(512 * 64 + 255) / 256, 256, 0, stream>>>(z, cw, sf);
  k_spat<<<(512 * 64 + 255) / 256, 256, 0, stream>>>(sf, sa_w, sa);
  k_apply<<<(512 * 64 * 128 + 255) / 256, 256, 0, stream>>>(z, cw, sa, znb);

  // ---- 1x1 projection to EMB_DIM (GEMM over 32768 positions) --------------
  cvt(pre_w, prewb, 64, 128, 128);
  gemm(znb, prewb, pre_b, zq, zqb, 32768, 64, 128, 0);

  // ---- vector quantizer ---------------------------------------------------
  cvt(codebook, cbb, 512, 64, 64);
  k_cnorm<<<2, 256, 0, stream>>>(codebook, cn);
  gemm(zqb, cbb, nullptr, (float*)big, nullptr, 32768, 512, 64, 0);   // dot products
  k_zero_hist<<<2, 256, 0, stream>>>(hist);
  k_argmin<<<32768 / 256, 256, 0, stream>>>((const float*)big, cn, vidx, hist);
  k_gather<<<(32768 * 64 + 255) / 256, 256, 0, stream>>>(vidx, codebook, quant);
  k_loss<<<8192, 256, 0, stream>>>(quant, zq, part);
  k_loss_final<<<1, 256, 0, stream>>>(part, out);                     // d_out[0] = loss
  k_perp<<<1, 512, 0, stream>>>(hist, out + (out_size - 1));          // d_out[last] = ppl

  // ---- decoder: 3x3 conv 64->128 via im2col + WMMA ------------------------
  k_repack3<<<(128 * 64 * 9 + 255) / 256, 256, 0, stream>>>(dec_conv1_w, dcwb, 128, 64);
  {
    long tot = 32768L * 576;
    k_im2col3<<<(unsigned)((tot + 255) / 256), 256, 0, stream>>>(quant, (u16*)big, 64, 0);
  }
  gemm((u16*)big, dcwb, dec_conv1_b, hdec, nullptr, 32768, 128, 576, 0);

  // ---- two residual blocks ------------------------------------------------
  for (int rb = 0; rb < 2; rb++) {
    const float* w1 = resp[rb * 6 + 0];
    const float* g1 = resp[rb * 6 + 1];
    const float* b1 = resp[rb * 6 + 2];
    const float* w2 = resp[rb * 6 + 3];
    const float* g2 = resp[rb * 6 + 4];
    const float* b2 = resp[rb * 6 + 5];
    k_repack3<<<(32 * 128 * 9 + 255) / 256, 256, 0, stream>>>(w1, rw1b, 32, 128);
    long tot = 32768L * 1152;
    k_im2col3<<<(unsigned)((tot + 255) / 256), 256, 0, stream>>>(hdec, (u16*)big, 128, 1);
    gemm((u16*)big, rw1b, nullptr, rbuf, nullptr, 32768, 32, 1152, 0);
    k_gn1<<<512 * 32, 64, 0, stream>>>(rbuf, g1, b1, rbufb);
    cvt(w2, rw2b, 128, 32, 32);
    gemm(rbufb, rw2b, nullptr, r2, nullptr, 32768, 128, 32, 0);
    k_gn2add<<<512 * 32, 256, 0, stream>>>(r2, g2, b2, hdec);
  }

  // ---- deconvs ------------------------------------------------------------
  k_deconv1<<<(512 * 16 * 16 * 64 + 255) / 256, 256, 0, stream>>>(hdec, deconv1_w, deconv1_b, dd1);
  k_deconv2<<<(512 * 3 * 32 * 32 + 255) / 256, 256, 0, stream>>>(dd1, deconv2_w, deconv2_b, out + 1);
}
